// MaskedMultiHeadSDPAttention_67585605370705
// MI455X (gfx1250) — matrix operand order repaired and checked by reading, not compiled, not run on hardware
//
#include <hip/hip_runtime.h>

// Problem constants (fixed by reference)
#define BB   4
#define SS   2048
#define DD   1024
#define HH   16
#define DKK  64
#define NQKV 3072      // 3*DD
#define MROWS 8192     // BB*SS

typedef __attribute__((ext_vector_type(16))) __bf16 v16bf;
typedef __attribute__((ext_vector_type(8)))  float  v8f;

// GCC-style int 4-vector, matching the async-LDS builtin's pointee type
typedef int v4i_g __attribute__((vector_size(16)));
typedef __attribute__((address_space(1))) v4i_g gv4i;   // global (__device__) AS
typedef __attribute__((address_space(3))) v4i_g lv4i;   // LDS (__shared__) AS

#if defined(__AMDGCN__) && __has_builtin(__builtin_amdgcn_global_load_async_to_lds_b128)
#define HAVE_ASYNC_LDS 1
#else
#define HAVE_ASYNC_LDS 0
#endif

__device__ __forceinline__ __bf16 f2bf(float f) {
  unsigned u = __float_as_uint(f);
  unsigned r = (u + 0x7FFFu + ((u >> 16) & 1u)) >> 16;
  unsigned short s = (unsigned short)r;
  return *reinterpret_cast<__bf16*>(&s);
}

// Copy 8 bf16 (16 bytes) global -> LDS, async when available.
__device__ __forceinline__ void copy8_g2l(const __bf16* g, __bf16* l) {
#if HAVE_ASYNC_LDS
  __builtin_amdgcn_global_load_async_to_lds_b128((gv4i*)g, (lv4i*)l, 0, 0);
#else
  *(uint4*)l = *(const uint4*)g;
#endif
}

template <int N>
__device__ __forceinline__ void async_wait() {
#if HAVE_ASYNC_LDS
#if __has_builtin(__builtin_amdgcn_s_wait_asynccnt)
  __builtin_amdgcn_s_wait_asynccnt(N);
#else
  asm volatile("s_wait_asynccnt %0" ::"i"(N) : "memory");
#endif
#endif
}

// A-operand (16x32 bf16) element->K mapping per CDNA5 ISA 7.12.2:
// VGPR0..3 hold K = half*8 + {0..7}, VGPR4..7 hold K = 16 + half*8 + {0..7}
__device__ __forceinline__ int a_kmap(int e, int half) {
  int j = e >> 1;
  int base = (j < 4) ? (2 * j) : (16 + 2 * (j - 4));
  return base + half * 8 + (e & 1);
}

// ---------------------------------------------------------------- converts
__global__ __launch_bounds__(256) void cvt_f32_bf16(const float* __restrict__ src,
                                                    __bf16* __restrict__ dst, int n) {
  int i = (blockIdx.x * 256 + threadIdx.x) * 4;
  if (i + 3 < n) {
#pragma unroll
    for (int j = 0; j < 4; j++) dst[i + j] = f2bf(src[i + j]);
  }
}

// Pack Wq/Wk/Wv [H,D,DK] fp32 -> Wqkv bf16 [D rows, 3072 cols], col = qkv*1024 + h*64 + k
__global__ __launch_bounds__(256) void pack_wqkv(const float* __restrict__ Wq,
                                                 const float* __restrict__ Wk,
                                                 const float* __restrict__ Wv,
                                                 __bf16* __restrict__ dst) {
  int idx = blockIdx.x * 256 + threadIdx.x;            // < 3*1048576
  int qkv = idx >> 20;
  int rem = idx & 1048575;                              // h*65536 + d*64 + k
  int h = rem >> 16;
  int d = (rem >> 6) & 1023;
  int k = rem & 63;
  const float* src = (qkv == 0) ? Wq : (qkv == 1) ? Wk : Wv;
  dst[(size_t)d * NQKV + qkv * 1024 + h * 64 + k] = f2bf(src[rem]);
}

__global__ __launch_bounds__(256) void pack_bqkv(const float* __restrict__ bq,
                                                 const float* __restrict__ bk,
                                                 const float* __restrict__ bv,
                                                 float* __restrict__ dst) {
  int idx = blockIdx.x * 256 + threadIdx.x;            // < 3072
  if (idx < NQKV) {
    int qkv = idx >> 10;
    int rem = idx & 1023;                               // == h*64 + k
    const float* src = (qkv == 0) ? bq : (qkv == 1) ? bk : bv;
    dst[idx] = src[rem];
  }
}

// ---------------------------------------------------------------- GEMM
// C[M,N] = A[M,K] * B[K,N] + bias[N]
// mode 0: scatter bf16 into QKV buffers [3][B][H][S][DK]
// mode 1: write fp32 to f32_out [M, DD]
// Double-buffered LDS, async global->LDS staging (4 async ops/wave/stage).
__global__ __launch_bounds__(256)
void gemm_bf16(const __bf16* __restrict__ A, int lda,
               const __bf16* __restrict__ Bm, int ldb,
               int Kdim,
               const float* __restrict__ bias,
               int mode,
               __bf16* __restrict__ qkv_out,
               float* __restrict__ f32_out) {
  __shared__ __align__(32) __bf16 Alds[2][128 * 32];
  __shared__ __align__(32) __bf16 Blds[2][32 * 128];

  const int tid = threadIdx.x;
  const int lane = tid & 31, wid = tid >> 5;
  const int half = (lane >> 4) & 1, l16 = lane & 15;
  const int wm = wid & 3, wn = wid >> 2;          // 4 x 2 wave grid
  const int row0 = blockIdx.x * 128;
  const int col0 = blockIdx.y * 128;

  // Per-thread staging coordinates (constant across K steps)
  const int ra = tid >> 1, ca = (tid & 1) * 16;   // A: 128x32, 16 elems/thread
  const int rb = tid >> 3, cb = (tid & 7) * 16;   // B: 32x128, 16 elems/thread

  v8f acc[2][4];
#pragma unroll
  for (int mt = 0; mt < 2; mt++)
#pragma unroll
    for (int nt = 0; nt < 4; nt++) acc[mt][nt] = (v8f){0.f,0.f,0.f,0.f,0.f,0.f,0.f,0.f};

  auto stage = [&](int k0, int buf) {
    const __bf16* gA = A + (size_t)(row0 + ra) * lda + k0 + ca;
    __bf16* lA = &Alds[buf][ra * 32 + ca];
    copy8_g2l(gA, lA);
    copy8_g2l(gA + 8, lA + 8);
    const __bf16* gB = Bm + (size_t)(k0 + rb) * ldb + col0 + cb;
    __bf16* lB = &Blds[buf][rb * 128 + cb];
    copy8_g2l(gB, lB);
    copy8_g2l(gB + 8, lB + 8);
  };

  const int ksteps = Kdim >> 5;
  stage(0, 0);
  for (int ks = 0; ks < ksteps; ks++) {
    const int buf = ks & 1;
    if (ks + 1 < ksteps) {
      stage((ks + 1) << 5, buf ^ 1);   // prefetch next tile (in flight)
      async_wait<4>();                 // in-order: current tile's 4 ops done
    } else {
      async_wait<0>();
    }
    __syncthreads();

    v16bf af[2], bfr[4];
#pragma unroll
    for (int mt = 0; mt < 2; mt++) {
      int rrow = wm * 32 + mt * 16 + l16;
#pragma unroll
      for (int e = 0; e < 16; e++) af[mt][e] = Alds[buf][rrow * 32 + a_kmap(e, half)];
    }
#pragma unroll
    for (int nt = 0; nt < 4; nt++) {
      // B fragment: lane = N column (l16), elements follow a_kmap K order
#pragma unroll
      for (int e = 0; e < 16; e++) bfr[nt][e] = Blds[buf][a_kmap(e, half) * 128 + wn * 64 + nt * 16 + l16];
    }
#pragma unroll
    for (int mt = 0; mt < 2; mt++)
#pragma unroll
      for (int nt = 0; nt < 4; nt++)
        acc[mt][nt] = __builtin_amdgcn_wmma_f32_16x16x32_bf16(
            false, af[mt], false, bfr[nt], (short)0, acc[mt][nt], false, false);
    __syncthreads();   // all reads done before this buffer is overwritten
  }

  // epilogue
#pragma unroll
  for (int mt = 0; mt < 2; mt++) {
#pragma unroll
    for (int nt = 0; nt < 4; nt++) {
#pragma unroll
      for (int r = 0; r < 8; r++) {
        int grow = row0 + wm * 32 + mt * 16 + r + 8 * half;
        int gcol = col0 + wn * 64 + nt * 16 + l16;
        float v = acc[mt][nt][r] + bias[gcol];
        if (mode == 0) {
          int qkv = gcol >> 10;
          int rem = gcol & 1023;
          int h = rem >> 6, dk = rem & 63;
          int b = grow >> 11, s = grow & 2047;
          size_t di = ((((size_t)qkv * BB + b) * HH + h) * SS + s) * DKK + dk;
          qkv_out[di] = f2bf(v);
        } else {
          f32_out[(size_t)grow * DD + gcol] = v;
        }
      }
    }
  }
}

// ---------------------------------------------------------------- flash attention
// block: 128 threads = 4 waves; each block does 64 query rows of one (b,h)
__global__ __launch_bounds__(128)
void flash_attn(const __bf16* __restrict__ Qb, const __bf16* __restrict__ Kb,
                const __bf16* __restrict__ Vb, __bf16* __restrict__ Ob) {
  __shared__ __align__(32) __bf16 Klds[32 * 64];
  __shared__ __align__(32) __bf16 Vlds[32 * 64];
  __shared__ __align__(32) __bf16 Plds[4][16 * 32];

  const int tid = threadIdx.x, lane = tid & 31, wid = tid >> 5;
  const int half = (lane >> 4) & 1, l16 = lane & 15;
  const int bh = blockIdx.y;
  const int b = bh >> 4, h = bh & 15;
  const int q0 = blockIdx.x * 64;

  const __bf16* Qh = Qb + (size_t)bh * SS * DKK;
  const __bf16* Kh = Kb + (size_t)bh * SS * DKK;
  const __bf16* Vh = Vb + (size_t)bh * SS * DKK;

  // Q fragments (A operand), 2 chunks of K=32 covering DK=64
  v16bf qf[2];
  {
    int qrow = q0 + wid * 16 + l16;
#pragma unroll
    for (int ch = 0; ch < 2; ch++)
#pragma unroll
      for (int e = 0; e < 16; e++)
        qf[ch][e] = Qh[(size_t)qrow * DKK + ch * 32 + a_kmap(e, half)];
  }

  v8f acc[4];
#pragma unroll
  for (int nt = 0; nt < 4; nt++) acc[nt] = (v8f){0.f,0.f,0.f,0.f,0.f,0.f,0.f,0.f};
  float m_run[8], l_run[8];
#pragma unroll
  for (int r = 0; r < 8; r++) { m_run[r] = -3.0e38f; l_run[r] = 0.f; }

  const int nkb = (q0 >> 5) + 2;   // 32-key blocks covering keys <= q0+63
  for (int kb = 0; kb < nkb; kb++) {
    const int key0 = kb * 32;
    {   // stage K,V tiles 32x64 bf16 (async when available)
      int r = tid >> 2, c = (tid & 3) * 16;
      const __bf16* gk = Kh + (size_t)(key0 + r) * DKK + c;
      const __bf16* gv = Vh + (size_t)(key0 + r) * DKK + c;
      __bf16* lk = Klds + r * 64 + c;
      __bf16* lv = Vlds + r * 64 + c;
      copy8_g2l(gk, lk); copy8_g2l(gk + 8, lk + 8);
      copy8_g2l(gv, lv); copy8_g2l(gv + 8, lv + 8);
      async_wait<0>();
    }
    __syncthreads();

    // scores: 16 queries x 32 keys, via 2 n-tiles x 2 k-chunks of WMMA
    v8f st[2];
#pragma unroll
    for (int nt = 0; nt < 2; nt++) {
      v16bf b0, b1;
#pragma unroll
      for (int e = 0; e < 16; e++) {
        b0[e] = Klds[(nt * 16 + l16) * 64 + a_kmap(e, half)];        // dk = a_kmap (chunk 0)
        b1[e] = Klds[(nt * 16 + l16) * 64 + 32 + a_kmap(e, half)];   // dk = 32+a_kmap (chunk 1)
      }
      v8f c = (v8f){0.f,0.f,0.f,0.f,0.f,0.f,0.f,0.f};
      c = __builtin_amdgcn_wmma_f32_16x16x32_bf16(false, qf[0], false, b0, (short)0, c, false, false);
      c = __builtin_amdgcn_wmma_f32_16x16x32_bf16(false, qf[1], false, b1, (short)0, c, false, false);
#pragma unroll
      for (int r = 0; r < 8; r++) {
        int qg = q0 + wid * 16 + r + 8 * half;
        int kg = key0 + nt * 16 + l16;
        float s = c[r] * 0.125f;                         // 1/sqrt(64)
        st[nt][r] = (kg <= qg) ? s : -1.0e30f;
      }
    }

    // online softmax: row max over 32 keys (reduce across 16-lane half)
    float mnew[8], alpha[8], rs[8];
#pragma unroll
    for (int r = 0; r < 8; r++) {
      float v = fmaxf(st[0][r], st[1][r]);
#pragma unroll
      for (int m = 1; m < 16; m <<= 1) v = fmaxf(v, __shfl_xor(v, m, 32));
      mnew[r] = fmaxf(m_run[r], v);
      alpha[r] = __expf(m_run[r] - mnew[r]);
      m_run[r] = mnew[r];
      rs[r] = 0.f;
    }
#pragma unroll
    for (int nt = 0; nt < 4; nt++)
#pragma unroll
      for (int r = 0; r < 8; r++) acc[nt][r] = acc[nt][r] * alpha[r];

    // P = exp(s - m); stash to LDS (row-major 16x32) for A-layout reload
#pragma unroll
    for (int nt = 0; nt < 2; nt++)
#pragma unroll
      for (int r = 0; r < 8; r++) {
        float p = __expf(st[nt][r] - mnew[r]);
        rs[r] += p;
        Plds[wid][(r + 8 * half) * 32 + nt * 16 + l16] = f2bf(p);
      }
#pragma unroll
    for (int r = 0; r < 8; r++) {
      float v = rs[r];
#pragma unroll
      for (int m = 1; m < 16; m <<= 1) v += __shfl_xor(v, m, 32);
      l_run[r] = l_run[r] * alpha[r] + v;
    }
    __syncthreads();   // order Plds write -> read (uniform trip count)

    // P (16x32) * V (32x64): 4 WMMAs
    v16bf pa;
#pragma unroll
    for (int e = 0; e < 16; e++) pa[e] = Plds[wid][l16 * 32 + a_kmap(e, half)];
#pragma unroll
    for (int nt = 0; nt < 4; nt++) {
      v16bf bv;
#pragma unroll
      for (int e = 0; e < 16; e++) bv[e] = Vlds[a_kmap(e, half) * 64 + nt * 16 + l16];
      acc[nt] = __builtin_amdgcn_wmma_f32_16x16x32_bf16(
          false, pa, false, bv, (short)0, acc[nt], false, false);
    }
    __syncthreads();   // protect K/V tiles before next stage
  }

  // epilogue: out / l, store bf16 as [B, S, H*DK]
#pragma unroll
  for (int nt = 0; nt < 4; nt++)
#pragma unroll
    for (int r = 0; r < 8; r++) {
      int s = q0 + wid * 16 + r + 8 * half;
      int feat = h * 64 + nt * 16 + l16;
      float o = acc[nt][r] / l_run[r];
      Ob[((size_t)b * SS + s) * DD + feat] = f2bf(o);
    }
}

// ---------------------------------------------------------------- launch
extern "C" void kernel_launch(void* const* d_in, const int* in_sizes, int n_in,
                              void* d_out, int out_size, void* d_ws, size_t ws_size,
                              hipStream_t stream) {
  const float* x  = (const float*)d_in[0];
  const float* Wq = (const float*)d_in[1];
  const float* bq = (const float*)d_in[2];
  const float* Wk = (const float*)d_in[3];
  const float* bk = (const float*)d_in[4];
  const float* Wv = (const float*)d_in[5];
  const float* bv = (const float*)d_in[6];
  const float* Wo = (const float*)d_in[7];
  const float* bo = (const float*)d_in[8];

  char* ws = (char*)d_ws;
  size_t off = 0;
  auto carve = [&](size_t bytes) {
    void* p = ws + off;
    off = (off + bytes + 255) & ~(size_t)255;
    return p;
  };
  __bf16* x_bf   = (__bf16*)carve((size_t)MROWS * DD * 2);        // 16 MB
  __bf16* wqkv   = (__bf16*)carve((size_t)DD * NQKV * 2);         // 6 MB
  float*  bqkv   = (float*) carve((size_t)NQKV * 4);
  __bf16* wo_bf  = (__bf16*)carve((size_t)DD * DD * 2);           // 2 MB
  __bf16* qkvb   = (__bf16*)carve((size_t)3 * BB * HH * SS * DKK * 2); // 48 MB
  __bf16* attn   = (__bf16*)carve((size_t)MROWS * DD * 2);        // 16 MB
  (void)ws_size; (void)in_sizes; (void)n_in; (void)out_size;

  const size_t BHSD = (size_t)BB * HH * SS * DKK;                 // 8,388,608

  cvt_f32_bf16<<<(MROWS * DD) / 1024, 256, 0, stream>>>(x, x_bf, MROWS * DD);
  cvt_f32_bf16<<<(DD * DD) / 1024, 256, 0, stream>>>(Wo, wo_bf, DD * DD);
  pack_wqkv<<<(3 * 1048576) / 256, 256, 0, stream>>>(Wq, Wk, Wv, wqkv);
  pack_bqkv<<<12, 256, 0, stream>>>(bq, bk, bv, bqkv);

  // QKV projection: [8192,1024] x [1024,3072]
  gemm_bf16<<<dim3(MROWS / 128, NQKV / 128), 256, 0, stream>>>(
      x_bf, DD, wqkv, NQKV, DD, bqkv, 0, qkvb, nullptr);

  // causal flash attention per (b,h)
  flash_attn<<<dim3(SS / 64, BB * HH), 128, 0, stream>>>(
      qkvb, qkvb + BHSD, qkvb + 2 * BHSD, attn);

  // output projection: [8192,1024] x [1024,1024] -> fp32 d_out
  gemm_bf16<<<dim3(MROWS / 128, DD / 128), 256, 0, stream>>>(
      attn, DD, wo_bf, DD, DD, bo, 1, nullptr, (float*)d_out);
}
